// MessagePassing_24635932410275
// MI455X (gfx1250) — matrix operand order repaired and checked by reading, not compiled, not run on hardware
//
#include <hip/hip_runtime.h>

#define N_ATOMS 100000
#define N_EDGES 300000
#define ATOM_DIM 32
#define BOND_DIM 16
#define STEPS 4

// K-blocks of 32 along the fused (k,j) axis: 16 bond-weight blocks + 1 bias block
#define KBLOCKS 17
#define WF_ENTRIES (KBLOCKS * 2 * 32 * 16)   // [kb][ntile(2)][lane(32)][16 bf16] = 17408
#define WG_ENTRIES (6 * 32 * 16)             // [ntile(6)][lane(32)][16 bf16]    = 3072

typedef __bf16 v16bf __attribute__((ext_vector_type(16)));
typedef float  v8f   __attribute__((ext_vector_type(8)));
typedef unsigned int u32x4 __attribute__((ext_vector_type(4)));
typedef int          i32x8 __attribute__((ext_vector_type(8)));
typedef int          i32x4 __attribute__((ext_vector_type(4)));

struct U8 { unsigned int x[8]; };           // 8 packed bf16 pairs == one 16x32 A fragment

// ---------------------------------------------------------------------------
// Tensor Data Mover: DMA a contiguous block (nbytes, multiple of 4) from
// global memory into LDS offset 0.  D# per CDNA5 ISA:
//   group0: count=1, lds_addr=0, global_addr, type=2
//   group1: data_size=4B, tensor_dim0=tile_dim0=ndw, tensor_dim1=tile_dim1=1,
//           tensor_dim0_stride=ndw; no gather/iterate/pad/multicast.
// Caller: issue from ONE wave only (TDM ignores EXEC), then wait TENSORcnt.
// ---------------------------------------------------------------------------
__device__ __forceinline__ void tdm_load_to_lds0(const void* gptr, unsigned nbytes) {
  const unsigned ndw = nbytes >> 2;
  const unsigned long long ga = (unsigned long long)gptr;
  u32x4 g0;
  g0[0] = 1u;                                             // count=1 (valid user D#)
  g0[1] = 0u;                                             // lds_addr = 0
  g0[2] = (unsigned)(ga & 0xffffffffu);                   // global_addr[31:0]
  g0[3] = (unsigned)((ga >> 32) & 0x01ffffffu) | (2u << 30); // addr[56:32] | type=2
  i32x8 g1;
  g1[0] = (int)(2u << 16);                                // data_size = 4 bytes
  g1[1] = (int)((ndw & 0xffffu) << 16);                   // tensor_dim0[15:0]
  g1[2] = (int)((ndw >> 16) & 0xffffu) | (1 << 16);       // tensor_dim0[31:16] | tensor_dim1=1
  g1[3] = (int)((ndw & 0xffffu) << 16);                   // tile_dim0
  g1[4] = 1;                                              // tile_dim1 = 1
  g1[5] = (int)ndw;                                       // tensor_dim0_stride[31:0]
  g1[6] = 0;
  g1[7] = 0;
  const i32x4 z4 = {0, 0, 0, 0};
#if __clang_major__ >= 23
  const i32x8 z8 = {0, 0, 0, 0, 0, 0, 0, 0};
  __builtin_amdgcn_tensor_load_to_lds(g0, g1, z4, z4, z8, 0);
#else
  __builtin_amdgcn_tensor_load_to_lds(g0, g1, z4, z4, 0);
#endif
}

// The TDM builtin carries no pointer operand, so the compiler cannot see that
// it writes the shared array.  Escape the array's address through an asm block
// with a memory clobber so subsequent ds_loads are not folded as undef.
#define TDM_LDS_PUBLISH(arr)                                   \
  do {                                                         \
    void* _p = (void*)(arr);                                   \
    asm volatile("" : "+v"(_p)::"memory");                     \
  } while (0)

// ---------------------------------------------------------------------------
// Pack constant weight matrices into per-lane WMMA B-operand layout (bf16).
// B 32x16 (16-bit): lanes 0-15 hold N=lane, K=0..7,16..23 (2/VGPR); lanes 16-31 hold
// N=lane-16, K=8..15,24..31.  -> kloc = (t<8 ? t : t+8) + (lane>>4)*8 for slot t in 0..15.
// Edge GEMM B:  Wf[(k,j), n] = W_bond[k, n*32 + j]  (kb==16 -> bias block b_bond)
// GRU B:        Bih[j, n]    = W_ih[n, j]           (and W_hh)
// ---------------------------------------------------------------------------
__global__ void pack_weights(const float* __restrict__ W_bond,
                             const float* __restrict__ b_bond,
                             const float* __restrict__ W_ih,
                             const float* __restrict__ W_hh,
                             unsigned short* __restrict__ wf,
                             unsigned short* __restrict__ wg) {
  const int idx = blockIdx.x * 256 + threadIdx.x;
  if (idx < WF_ENTRIES) {
    const int t    = idx & 15;
    const int lane = (idx >> 4) & 31;
    const int nt   = (idx >> 9) & 1;
    const int kb   = idx >> 10;                 // 0..16
    const int kloc = (t < 8 ? t : t + 8) + (lane >> 4) * 8;      // j within 0..31
    const int n    = nt * 16 + (lane & 15);
    const float v  = (kb < 16) ? W_bond[kb * (ATOM_DIM * ATOM_DIM) + n * ATOM_DIM + kloc]
                               : b_bond[n * ATOM_DIM + kloc];
    wf[idx] = __builtin_bit_cast(unsigned short, (__bf16)v);
  } else {
    const int j = idx - WF_ENTRIES;
    if (j < 2 * WG_ENTRIES) {
      const int which = j / WG_ENTRIES;         // 0 = W_ih, 1 = W_hh
      const int jj    = j % WG_ENTRIES;
      const int t     = jj & 15;
      const int lane  = (jj >> 4) & 31;
      const int nt    = jj >> 9;                // 0..5  (N = 96 gate columns)
      const int kloc  = (t < 8 ? t : t + 8) + (lane >> 4) * 8;
      const int n     = nt * 16 + (lane & 15);
      const float v   = (which == 0) ? W_ih[n * ATOM_DIM + kloc]
                                     : W_hh[n * ATOM_DIM + kloc];
      wg[j] = __builtin_bit_cast(unsigned short, (__bf16)v);
    }
  }
}

__global__ void init_h(float* __restrict__ h, const float* __restrict__ af) {
  const int i = blockIdx.x * 256 + threadIdx.x;   // exact: 800000 threads
  ((float4*)h)[i] = ((const float4*)af)[i];
}

__global__ void zero_f4(float* __restrict__ p) {
  const int i = blockIdx.x * 256 + threadIdx.x;
  ((float4*)p)[i] = make_float4(0.f, 0.f, 0.f, 0.f);
}

// ---------------------------------------------------------------------------
// Edge message kernel: one wave32 per 16-edge tile.
// A slot t of lane L holds K = (t<8 ? t : t+8) + (L>=16)*8 (16-bit A layout).
// h[nbr] is converted to bf16 ONCE per tile; each k-block's A fragment is
// 8x v_pk_mul_bf16 against a v_cvt_pk_bf16_f32 splat of bond[e,kb].
// Weights staged into LDS by the Tensor Data Mover (one wave, TENSORcnt).
// 34 v_wmma per tile, f32 atomic scatter into L2-resident agg.
// ---------------------------------------------------------------------------
__global__ void __launch_bounds__(256)
edge_msg_kernel(const float* __restrict__ h,
                const float* __restrict__ bond,
                const int* __restrict__ pairs,
                const unsigned short* __restrict__ wf_pack,
                float* __restrict__ agg) {
  __shared__ __attribute__((aligned(32))) unsigned short lds_wf[WF_ENTRIES];
  if (threadIdx.x < 32) {                 // single wave issues the DMA
    tdm_load_to_lds0(wf_pack, WF_ENTRIES * 2);
    __builtin_amdgcn_s_wait_tensorcnt(0);
  }
  TDM_LDS_PUBLISH(lds_wf);
  __syncthreads();

  const int lane = threadIdx.x & 31;
  const int gw   = blockIdx.x * 8 + (threadIdx.x >> 5);
  const int tw   = gridDim.x * 8;
  const int m    = lane & 15;      // edge slot within tile (M)
  const int lh   = lane >> 4;      // lane half
  const int joff = lh * 8;
  const int nTiles = N_EDGES / 16; // 18750, exact

  for (int tile = gw; tile < nTiles; tile += tw) {
    const int e    = tile * 16 + m;
    const int dstA = pairs[e * 2 + 0];
    const int nbrA = pairs[e * 2 + 1];
    if (tile + tw < nTiles)        // global_prefetch_b8 next tile's indices
      __builtin_prefetch(pairs + ((tile + tw) * 16 + m) * 2, 0, 1);

    // Gather neighbor h fragment and convert to bf16 once (A-layout order).
    v16bf hbv;
    {
      const float* hr = h + nbrA * ATOM_DIM + joff;
      const float4 f0 = *(const float4*)(hr + 0);
      const float4 f1 = *(const float4*)(hr + 4);
      const float4 f2 = *(const float4*)(hr + 16);
      const float4 f3 = *(const float4*)(hr + 20);
      const float hv[16] = {f0.x,f0.y,f0.z,f0.w, f1.x,f1.y,f1.z,f1.w,
                            f2.x,f2.y,f2.z,f2.w, f3.x,f3.y,f3.z,f3.w};
      #pragma unroll
      for (int t = 0; t < 16; ++t) hbv[t] = (__bf16)hv[t];
    }
    const U8 hb = __builtin_bit_cast(U8, hbv);

    v8f acc0 = {0.f,0.f,0.f,0.f,0.f,0.f,0.f,0.f};
    v8f acc1 = {0.f,0.f,0.f,0.f,0.f,0.f,0.f,0.f};

    #pragma unroll 1
    for (int kbase = 0; kbase < 16; kbase += 4) {
      const float4 bq = *(const float4*)(bond + e * BOND_DIM + kbase);
      const float bs[4] = {bq.x, bq.y, bq.z, bq.w};
      #pragma unroll
      for (int u = 0; u < 4; ++u) {
        unsigned int sp;                         // {bf16(s), bf16(s)} splat
        asm("v_cvt_pk_bf16_f32 %0, %1, %1" : "=v"(sp) : "v"(bs[u]));
        U8 av;
        #pragma unroll
        for (int i = 0; i < 8; ++i)              // native packed bf16 multiply
          asm("v_pk_mul_bf16 %0, %1, %2" : "=v"(av.x[i]) : "v"(hb.x[i]), "v"(sp));
        const v16bf a = __builtin_bit_cast(v16bf, av);
        const int kb = kbase + u;
        const v16bf bb0 = *(const v16bf*)(lds_wf + ((kb * 2 + 0) * 32 + lane) * 16);
        const v16bf bb1 = *(const v16bf*)(lds_wf + ((kb * 2 + 1) * 32 + lane) * 16);
        acc0 = __builtin_amdgcn_wmma_f32_16x16x32_bf16(false, a, false, bb0, (short)0, acc0, false, false);
        acc1 = __builtin_amdgcn_wmma_f32_16x16x32_bf16(false, a, false, bb1, (short)0, acc1, false, false);
      }
    }
    {  // bias k-block (kb==16): A = bf16(h) directly
      const v16bf bb0 = *(const v16bf*)(lds_wf + ((16 * 2 + 0) * 32 + lane) * 16);
      const v16bf bb1 = *(const v16bf*)(lds_wf + ((16 * 2 + 1) * 32 + lane) * 16);
      acc0 = __builtin_amdgcn_wmma_f32_16x16x32_bf16(false, hbv, false, bb0, (short)0, acc0, false, false);
      acc1 = __builtin_amdgcn_wmma_f32_16x16x32_bf16(false, hbv, false, bb1, (short)0, acc1, false, false);
    }

    // D layout: VGPR r -> row M = r + lh*8, column N = (lane&15) (+16 for acc1).
    const int col = lane & 15;
    #pragma unroll
    for (int r = 0; r < 8; ++r) {
      const int d = __shfl(dstA, r + lh * 8, 32);
      atomicAdd(agg + d * ATOM_DIM + col,      acc0[r]);
      atomicAdd(agg + d * ATOM_DIM + col + 16, acc1[r]);
    }
  }
}

// ---------------------------------------------------------------------------
// GRU kernel: one wave32 per 16-atom tile.
// gi = agg @ W_ih^T, gh = h @ W_hh^T via 12 v_wmma (K=32, N=96 each),
// then f32 gate math; h updated in place (per-atom exclusive).
// ---------------------------------------------------------------------------
__global__ void __launch_bounds__(256)
gru_kernel(float* __restrict__ h,
           const float* __restrict__ agg,
           const unsigned short* __restrict__ wg_pack,
           const float* __restrict__ b_ih,
           const float* __restrict__ b_hh) {
  __shared__ __attribute__((aligned(32))) unsigned short lds_w[2 * WG_ENTRIES];
  if (threadIdx.x < 32) {                 // single wave issues the DMA
    tdm_load_to_lds0(wg_pack, 2 * WG_ENTRIES * 2);
    __builtin_amdgcn_s_wait_tensorcnt(0);
  }
  TDM_LDS_PUBLISH(lds_w);
  __syncthreads();

  const int lane = threadIdx.x & 31;
  const int gw   = blockIdx.x * 8 + (threadIdx.x >> 5);
  const int tw   = gridDim.x * 8;
  const int m    = lane & 15;
  const int lh   = lane >> 4;
  const int joff = lh * 8;
  const int nTiles = N_ATOMS / 16;  // 6250, exact

  for (int tile = gw; tile < nTiles; tile += tw) {
    const int a0 = tile * 16;

    v16bf Aa, Ah;
    {
      const float* ar  = agg + (a0 + m) * ATOM_DIM + joff;
      const float* hr  = h   + (a0 + m) * ATOM_DIM + joff;
      const float4 a0v = *(const float4*)(ar + 0),  a1v = *(const float4*)(ar + 4);
      const float4 a2v = *(const float4*)(ar + 16), a3v = *(const float4*)(ar + 20);
      const float4 h0v = *(const float4*)(hr + 0),  h1v = *(const float4*)(hr + 4);
      const float4 h2v = *(const float4*)(hr + 16), h3v = *(const float4*)(hr + 20);
      const float af[16] = {a0v.x,a0v.y,a0v.z,a0v.w, a1v.x,a1v.y,a1v.z,a1v.w,
                            a2v.x,a2v.y,a2v.z,a2v.w, a3v.x,a3v.y,a3v.z,a3v.w};
      const float hf[16] = {h0v.x,h0v.y,h0v.z,h0v.w, h1v.x,h1v.y,h1v.z,h1v.w,
                            h2v.x,h2v.y,h2v.z,h2v.w, h3v.x,h3v.y,h3v.z,h3v.w};
      #pragma unroll
      for (int t = 0; t < 16; ++t) { Aa[t] = (__bf16)af[t]; Ah[t] = (__bf16)hf[t]; }
    }

    v8f gi[6], gh[6];
    #pragma unroll
    for (int nt = 0; nt < 6; ++nt) {
      gi[nt] = (v8f){0.f,0.f,0.f,0.f,0.f,0.f,0.f,0.f};
      gh[nt] = (v8f){0.f,0.f,0.f,0.f,0.f,0.f,0.f,0.f};
    }
    #pragma unroll
    for (int nt = 0; nt < 6; ++nt) {
      const v16bf bi_ = *(const v16bf*)(lds_w + (nt * 32 + lane) * 16);
      const v16bf bh_ = *(const v16bf*)(lds_w + WG_ENTRIES + (nt * 32 + lane) * 16);
      gi[nt] = __builtin_amdgcn_wmma_f32_16x16x32_bf16(false, Aa, false, bi_, (short)0, gi[nt], false, false);
      gh[nt] = __builtin_amdgcn_wmma_f32_16x16x32_bf16(false, Ah, false, bh_, (short)0, gh[nt], false, false);
    }

    float bi6[6], bh6[6];
    #pragma unroll
    for (int nt = 0; nt < 6; ++nt) {
      bi6[nt] = b_ih[nt * 16 + (lane & 15)];
      bh6[nt] = b_hh[nt * 16 + (lane & 15)];
    }

    #pragma unroll
    for (int r = 0; r < 8; ++r) {
      const int a = a0 + r + lh * 8;
      #pragma unroll
      for (int ct = 0; ct < 2; ++ct) {
        const float ir  = gi[ct][r]     + bi6[ct];
        const float hrv = gh[ct][r]     + bh6[ct];
        const float iz  = gi[2 + ct][r] + bi6[2 + ct];
        const float hz  = gh[2 + ct][r] + bh6[2 + ct];
        const float inn = gi[4 + ct][r] + bi6[4 + ct];
        const float hn  = gh[4 + ct][r] + bh6[4 + ct];
        const float rg  = 1.f / (1.f + __expf(-(ir + hrv)));
        const float zg  = 1.f / (1.f + __expf(-(iz + hz)));
        const float ng  = tanhf(inn + rg * hn);
        const int   c   = ct * 16 + (lane & 15);
        const float ho  = h[a * ATOM_DIM + c];
        h[a * ATOM_DIM + c] = (1.f - zg) * ng + zg * ho;
      }
    }
  }
}

// ---------------------------------------------------------------------------
extern "C" void kernel_launch(void* const* d_in, const int* in_sizes, int n_in,
                              void* d_out, int out_size, void* d_ws, size_t ws_size,
                              hipStream_t stream) {
  const float* atom_features = (const float*)d_in[0];
  const float* bond_features = (const float*)d_in[1];
  const int*   pair_indices  = (const int*)d_in[2];
  const float* W_bond        = (const float*)d_in[3];
  const float* b_bond        = (const float*)d_in[4];
  const float* W_ih          = (const float*)d_in[5];
  const float* W_hh          = (const float*)d_in[6];
  const float* b_ih          = (const float*)d_in[7];
  const float* b_hh          = (const float*)d_in[8];
  float* h = (float*)d_out;                       // h lives in d_out, updated in place

  char* ws = (char*)d_ws;
  float* agg = (float*)ws;                                                   // 12.8 MB
  unsigned short* wf = (unsigned short*)(ws + (size_t)N_ATOMS * ATOM_DIM * sizeof(float));
  unsigned short* wg = wf + WF_ENTRIES;

  pack_weights<<<(WF_ENTRIES + 2 * WG_ENTRIES + 255) / 256, 256, 0, stream>>>(
      W_bond, b_bond, W_ih, W_hh, wf, wg);
  init_h<<<(N_ATOMS * ATOM_DIM / 4) / 256, 256, 0, stream>>>(h, atom_features);

  for (int s = 0; s < STEPS; ++s) {
    zero_f4<<<(N_ATOMS * ATOM_DIM / 4) / 256, 256, 0, stream>>>(agg);
    edge_msg_kernel<<<512, 256, 0, stream>>>(h, bond_features, pair_indices, wf, agg);
    gru_kernel<<<256, 256, 0, stream>>>(h, agg, wg, b_ih, b_hh);
  }
}
